// TMP2_SMorphLayer_4037269258264
// MI455X (gfx1250) — matrix operand order repaired and checked by reading, not compiled, not run on hardware
//
#include <hip/hip_runtime.h>

// SMorph layer, refactored as GEMM for CDNA5 WMMA.
//
// smooth_max identity:  e^{x+k} = e^x * e^k, so per pixel we build features
//   e1 = e^x, f1 = x e^x, e2 = e^{-x}, f2 = -x e^{-x}   (75 each -> K=300, pad 320)
// and per weight constants E = e^k, KE = k e^k, giving per channel c:
//   num1 = f1.E1_c + e1.KE1_c ; den1 = e1.E1_c ; (branch 2 analogous)
//   out  = num1/den1 + num2/den2 + bias_c
// -> one GEMM: M = 96,800 pixels, K = 320, N = 64 (16 ch x {num1,den1,num2,den2})
// Memory-bound: 29 MB in / 6.2 MB out @ 23.3 TB/s; GEMM flops ~4 GF (trivial).
//
// v2 changes (from disasm feedback):
//  - LDS row stride padded 320 -> 328 f16: row base banks go from {0,32} only
//    (8-way conflicts) to a perfect 2-per-bank spread for ds_load_b128.
//  - __syncthreads() -> wave-local `s_wait_dscnt 0` (A panel is same-wave
//    produce/consume; waves run decoupled).
//  - x streaming vectorized to global_load_b128 (4 pixels per load);
//    epilogue stores packed into 2x global_store_b128 per lane.

typedef __attribute__((ext_vector_type(16))) _Float16 v16h;
typedef __attribute__((ext_vector_type(8)))  _Float16 v8h;
typedef __attribute__((ext_vector_type(8)))  float    v8f;
typedef __attribute__((ext_vector_type(4)))  float    v4f;

#define COUT   16
#define PATCH  75
#define HW     48400          // 220*220
#define TOTAL_PIX 96800       // 2*HW ; HW % 16 == 0 -> tiles never straddle batch
#define KDIM   320            // 4*75 padded to multiple of 32 (logical K)
#define LDS_STRIDE 328        // padded row stride: 164 dwords, 164%64=36 -> conflict-free
#define KSTEPS 10             // KDIM/32
#define NTILES 4              // N = 64 -> 4 tiles of 16
#define WAVES_PER_BLOCK 4
#define BLOCK_THREADS   128
#define PIX_PER_WAVE    16

// ---------------------------------------------------------------------------
// Prep: bake the 320x64 weight matrix directly into per-lane WMMA B-fragment
// layout (dense f16 B 32x16: lanes 0-15 -> N=lane, K=0..7,16..23; lanes 16-31 ->
// N=lane-16, K=8..15,24..31; matches the A fragment K order).
// bfrag layout: [kstep][ntile][lane][j], f16. Total 40 KB in d_ws.
// ---------------------------------------------------------------------------
__global__ void smorph_prep(const float* __restrict__ k1,
                            const float* __restrict__ k2,
                            _Float16* __restrict__ bfrag) {
  const int total = KSTEPS * NTILES * 32 * 16;
  for (int idx = threadIdx.x; idx < total; idx += blockDim.x) {
    int j     = idx & 15;
    int l     = (idx >> 4) & 31;
    int frag  = idx >> 9;
    int nt    = frag & 3;               // 0:num1 1:den1 2:num2 3:den2
    int kstep = frag >> 2;
    int k = kstep * 32 + ((j >> 3) * 16) + ((l >> 4) * 8) + (j & 7);   // B fragment K mapping
    int c = l & 15;                              // output channel
    float v = 0.0f;
    if (k < 4 * PATCH) {
      int feat = k / PATCH;             // 0:e1 1:f1 2:e2 3:f2
      int p    = k - feat * PATCH;
      if (nt == 0) {                    // num1 = f1.E1 + e1.KE1
        float kv = k1[p * COUT + c];
        if (feat == 0)      v = kv * __expf(kv);   // pairs with e1
        else if (feat == 1) v = __expf(kv);        // pairs with f1
      } else if (nt == 1) {             // den1 = e1.E1
        if (feat == 0)      v = __expf(k1[p * COUT + c]);
      } else if (nt == 2) {             // num2 = f2.E2 + e2.KE2
        float kv = k2[p * COUT + c];
        if (feat == 2)      v = kv * __expf(kv);
        else if (feat == 3) v = __expf(kv);
      } else {                          // den2 = e2.E2
        if (feat == 2)      v = __expf(k2[p * COUT + c]);
      }
    }
    bfrag[idx] = (_Float16)v;
  }
}

// ---------------------------------------------------------------------------
// Main: one wave = 16 pixels, fully wave-independent. Stage 16x320 f16 A
// panel in LDS (padded rows), 10 K-steps x 4 N-tiles of
// v_wmma_f32_16x16x32_f16, divide + bias epilogue with packed b128 stores.
// ---------------------------------------------------------------------------
__global__ void __launch_bounds__(BLOCK_THREADS)
smorph_main(const float* __restrict__ x,
            const _Float16* __restrict__ bfrag,
            const float* __restrict__ bias,
            float* __restrict__ out) {
  __shared__ __align__(16) _Float16 lds[WAVES_PER_BLOCK][PIX_PER_WAVE * LDS_STRIDE];

  const int lane  = threadIdx.x & 31;
  const int wslot = threadIdx.x >> 5;
  const int gwave = blockIdx.x * WAVES_PER_BLOCK + wslot;
  const int pixel_base = gwave * PIX_PER_WAVE;
  if (pixel_base >= TOTAL_PIX) return;          // wave-uniform; EXEC stays full

  const int bb  = pixel_base / HW;              // uniform across the tile
  const int hwb = pixel_base - bb * HW;
  const float* xb = x + (size_t)bb * PATCH * HW + hwb;
  _Float16* A = &lds[wslot][0];

  // zero the K padding (k = 300..319 for all 16 pixels)
  for (int idx = lane; idx < PIX_PER_WAVE * (KDIM - 4 * PATCH); idx += 32) {
    int m = idx & 15;
    int k = 4 * PATCH + (idx >> 4);
    A[m * LDS_STRIDE + k] = (_Float16)0.0f;
  }

  // Stream x with b128 loads: idx = p*4 + g -> lane loads pixels 4g..4g+3 of
  // patch row p (16-float tile base is 64B aligned; HW % 16 == 0).
  for (int idx = lane; idx < PATCH * 4; idx += 32) {
    int g = idx & 3;
    int p = idx >> 2;
    v4f v = *(const v4f*)(xb + (size_t)p * HW + 4 * g);
#pragma unroll
    for (int e = 0; e < 4; ++e) {
      int m = 4 * g + e;
      float vv = v[e];
      float e1 = __expf(vv);
      float e2 = __expf(-vv);
      _Float16* row = A + m * LDS_STRIDE;
      row[p]             = (_Float16)e1;
      row[PATCH + p]     = (_Float16)(vv * e1);
      row[2 * PATCH + p] = (_Float16)e2;
      row[3 * PATCH + p] = (_Float16)(-vv * e2);
    }
  }

  // Wave-local LDS fence: producer and consumer are the same wave (wave32
  // lockstep), so only DScnt needs draining -- no block barrier.
  asm volatile("s_wait_dscnt 0x0" ::: "memory");

  v8f acc[NTILES] = {};

  // A fragment per lane: M = lane&15; K runs [off, off+8) and [off+16, off+24)
  // relative to the k-step base -> two contiguous 16B LDS loads, conflict-free
  // with the padded row stride.
  const int m   = lane & 15;
  const int off = (lane >> 4) * 8;
  const _Float16* Arow = A + m * LDS_STRIDE;

#pragma unroll
  for (int ks = 0; ks < KSTEPS; ++ks) {
    const int kbase = ks * 32;
    v8h alo = *(const v8h*)(Arow + kbase + off);
    v8h ahi = *(const v8h*)(Arow + kbase + off + 16);
    v16h a  = __builtin_shufflevector(alo, ahi,
                                      0, 1, 2, 3, 4, 5, 6, 7,
                                      8, 9, 10, 11, 12, 13, 14, 15);
#pragma unroll
    for (int nt = 0; nt < NTILES; ++nt) {
      const _Float16* bp = bfrag + (size_t)(ks * NTILES + nt) * 512 + lane * 16;
      v16h bf = *(const v16h*)bp;                // 32 B/lane, L2-resident
      acc[nt] = __builtin_amdgcn_wmma_f32_16x16x32_f16(
          false, a, false, bf, (short)0, acc[nt], false, false);
    }
  }

  // Epilogue: C layout -> lane holds (M = r + 8*(lane>>4), N = lane&15).
  // The 8 results per lane are contiguous in hw -> two b128 stores.
  const int ch  = lane & 15;
  const float bs = bias[ch];
  float* outc = out + (size_t)(bb * COUT + ch) * HW + hwb + 8 * (lane >> 4);

  v4f y0, y1;
#pragma unroll
  for (int r = 0; r < 4; ++r)
    y0[r] = acc[0][r] / acc[1][r] + acc[2][r] / acc[3][r] + bs;
#pragma unroll
  for (int r = 4; r < 8; ++r)
    y1[r - 4] = acc[0][r] / acc[1][r] + acc[2][r] / acc[3][r] + bs;

  *(v4f*)(outc)     = y0;
  *(v4f*)(outc + 4) = y1;
}

// ---------------------------------------------------------------------------
extern "C" void kernel_launch(void* const* d_in, const int* in_sizes, int n_in,
                              void* d_out, int out_size, void* d_ws, size_t ws_size,
                              hipStream_t stream) {
  (void)in_sizes; (void)n_in; (void)out_size; (void)ws_size;
  const float* x    = (const float*)d_in[0];
  const float* k1   = (const float*)d_in[1];
  const float* k2   = (const float*)d_in[2];
  const float* bias = (const float*)d_in[3];
  float* out        = (float*)d_out;
  _Float16* bfrag   = (_Float16*)d_ws;          // 40 KB needed

  smorph_prep<<<1, 256, 0, stream>>>(k1, k2, bfrag);

  const int waves  = TOTAL_PIX / PIX_PER_WAVE;                        // 6050
  const int blocks = (waves + WAVES_PER_BLOCK - 1) / WAVES_PER_BLOCK; // 1513
  smorph_main<<<blocks, BLOCK_THREADS, 0, stream>>>(x, bfrag, bias, out);
}
